// PhysBiformer_54898271977481
// MI455X (gfx1250) — matrix operand order repaired and checked by reading, not compiled, not run on hardware
//
#include <hip/hip_runtime.h>
#include <hip/hip_bf16.h>

// ---------------- types for WMMA ----------------
typedef __attribute__((ext_vector_type(16))) _Float16 v16h;
typedef __attribute__((ext_vector_type(8)))  _Float16 v8h;
typedef __attribute__((ext_vector_type(8)))  float    v8f;

// ---------------- model constants ----------------
#define C_DIM   64
#define H_DIM   256
#define LTOK    10240      // 40*16*16
#define T_SNN   4
#define M_ROWS  (T_SNN*LTOK)   // 40960
#define N_WINWT 10
#define WINS    64         // 4*4*4 windows
#define SWIN    160        // 10*4*4 tokens per window
#define KGLEN   640        // TOPK * SWIN
#define EPSF    1e-5f

// ============================================================
// WMMA fragment helpers (layouts per CDNA5 ISA 7.12.2)
// ============================================================
__device__ inline v16h load_a_frag(const _Float16* __restrict__ A, int lda,
                                   int row0, int k0, int lane) {
    int m  = lane & 15;
    int ko = (lane >> 4) << 3;              // 0 or 8
    const _Float16* base = A + (size_t)(row0 + m) * lda + k0 + ko;
    v8h lo = *(const v8h*)(base);           // K = k0+ko .. +7
    v8h hi = *(const v8h*)(base + 16);      // K = k0+16+ko .. +7
    v16h a;
#pragma unroll
    for (int i = 0; i < 8; ++i) { a[i] = lo[i]; a[i + 8] = hi[i]; }
    return a;
}

__device__ inline v16h load_b_frag(const _Float16* __restrict__ Bt, int ldb,
                                   int col0, int k0, int lane) {
    int n  = lane & 15;
    int ko = (lane >> 4) << 3;              // 0 or 8
    const _Float16* base = Bt + (size_t)(col0 + n) * ldb + k0 + ko;
    v8h lo = *(const v8h*)(base);
    v8h hi = *(const v8h*)(base + 16);
    v16h b;
#pragma unroll
    for (int i = 0; i < 8; ++i) { b[i] = lo[i]; b[i + 8] = hi[i]; }
    return b;
}

// window forward map: token -> (w, s)
__device__ inline void tok_to_ws(int tok, int& w, int& s) {
    int lt = tok >> 8, lh = (tok >> 4) & 15, lw = tok & 15;
    w = ((lt / N_WINWT) * 4 + (lh >> 2)) * 4 + (lw >> 2);
    s = ((lt % N_WINWT) * 4 + (lh & 3)) * 4 + (lw & 3);
}
// window inverse map: (w, s) -> token
__device__ inline int ws_to_tok(int w, int s) {
    int lt = (w >> 4) * N_WINWT + (s >> 4);
    int lh = ((w >> 2) & 3) * 4 + ((s >> 2) & 3);
    int lw = (w & 3) * 4 + (s & 3);
    return (lt * 16 + lh) * 16 + lw;
}

// ============================================================
// Generic WMMA GEMM: D = A[MxK](f16,row) * Bt[NxK](f16,row)^T + bias
// MODE 0: D[m*N+n]
// MODE 1: rows are (t*LTOK+tok); store windowed qkv layout [(n*64+w)*160+s][N]
// MODE 2: rows are (t*LTOK + w*160+s); store D[(t*64+col)*LTOK+tok] = Res + val
// MODE 3: rows are (t*LTOK+tok); store D[(t*N+col)*LTOK+tok]
// ============================================================
template<int MODE>
__global__ void __launch_bounds__(128)
gemm_wmma(const _Float16* __restrict__ A, const _Float16* __restrict__ Bt,
          const float* __restrict__ bias, float* __restrict__ D,
          const float* __restrict__ Res, int M, int N, int K) {
    int wave = blockIdx.x * (blockDim.x >> 5) + (threadIdx.x >> 5);
    int lane = threadIdx.x & 31;
    int mt = M >> 5, nt = N >> 5;
    if (wave >= mt * nt) return;
    int tm = (wave / nt) << 5;
    int tn = (wave % nt) << 5;

    v8f c00 = {}, c01 = {}, c10 = {}, c11 = {};
    for (int k0 = 0; k0 < K; k0 += 64) {
#pragma unroll
        for (int kk = 0; kk < 2; ++kk) {
            int k = k0 + kk * 32;
            v16h a0 = load_a_frag(A, K, tm,      k, lane);
            v16h a1 = load_a_frag(A, K, tm + 16, k, lane);
            v16h b0 = load_b_frag(Bt, K, tn,      k, lane);
            v16h b1 = load_b_frag(Bt, K, tn + 16, k, lane);
            c00 = __builtin_amdgcn_wmma_f32_16x16x32_f16(false, a0, false, b0, (short)0, c00, false, false);
            c01 = __builtin_amdgcn_wmma_f32_16x16x32_f16(false, a0, false, b1, (short)0, c01, false, false);
            c10 = __builtin_amdgcn_wmma_f32_16x16x32_f16(false, a1, false, b0, (short)0, c10, false, false);
            c11 = __builtin_amdgcn_wmma_f32_16x16x32_f16(false, a1, false, b1, (short)0, c11, false, false);
        }
    }

    int nn   = lane & 15;
    int halfw = lane >> 4;
    v8f frg[4] = { c00, c01, c10, c11 };
#pragma unroll
    for (int f = 0; f < 4; ++f) {
        int row0 = tm + (f >> 1) * 16;
        int col  = tn + (f & 1) * 16 + nn;
        float bv = bias ? bias[col] : 0.0f;
#pragma unroll
        for (int v = 0; v < 8; ++v) {
            int m = row0 + v + halfw * 8;
            float val = frg[f][v] + bv;
            if (MODE == 0) {
                D[(size_t)m * N + col] = val;
            } else if (MODE == 1) {
                int nidx = m / LTOK, tok = m % LTOK;
                int w, s; tok_to_ws(tok, w, s);
                D[((size_t)((nidx * WINS + w) * SWIN + s)) * N + col] = val;
            } else if (MODE == 2) {
                int nidx = m / LTOK, r = m % LTOK;
                int w = r / SWIN, s = r % SWIN;
                int tok = ws_to_tok(w, s);
                size_t o = ((size_t)(nidx * C_DIM + col)) * LTOK + tok;
                D[o] = Res[o] + val;
            } else {   // MODE 3
                int tt = m / LTOK, tok = m % LTOK;
                D[((size_t)(tt * N + col)) * LTOK + tok] = val;
            }
        }
    }
}

// ============================================================
// BatchNorm statistics (training mode, biased variance)
// x laid out as [T][C][L]; one block per channel
// ============================================================
__global__ void __launch_bounds__(256)
bn_stats(const float* __restrict__ x, float* __restrict__ mean,
         float* __restrict__ istd, int C, int L, int T) {
    int c = blockIdx.x;
    float s = 0.f, ss = 0.f;
    for (int t = 0; t < T; ++t) {
        const float* p = x + ((size_t)t * C + c) * L;
        for (int i = threadIdx.x; i < L; i += blockDim.x) {
            float v = p[i]; s += v; ss += v * v;
        }
    }
    __shared__ float rs[256], rq[256];
    rs[threadIdx.x] = s; rq[threadIdx.x] = ss;
    __syncthreads();
    for (int o = 128; o > 0; o >>= 1) {
        if (threadIdx.x < o) { rs[threadIdx.x] += rs[threadIdx.x + o]; rq[threadIdx.x] += rq[threadIdx.x + o]; }
        __syncthreads();
    }
    if (threadIdx.x == 0) {
        float cnt = (float)L * (float)T;
        float m = rs[0] / cnt;
        float v = rq[0] / cnt - m * m;
        mean[c] = m;
        istd[c] = rsqrtf(v + EPSF);
    }
}

// ============================================================
// fused BN-apply + multistep LIF (tau=2, Vth=0.5, soft reset)
// x [T][C][L] f32 -> spikes in GEMM-A layout [(t*L+l)][C] f16
// ============================================================
__global__ void bn_lif(const float* __restrict__ x, const float* __restrict__ mean,
                       const float* __restrict__ istd, const float* __restrict__ g,
                       const float* __restrict__ b, _Float16* __restrict__ sa,
                       int C, int L) {
    int i = blockIdx.x * blockDim.x + threadIdx.x;
    if (i >= C * L) return;
    int c = i / L, l = i % L;
    float sc = g[c] * istd[c];
    float of = b[c] - mean[c] * sc;
    float v = 0.f;
    for (int t = 0; t < T_SNN; ++t) {
        float xn = x[((size_t)t * C + c) * L + l] * sc + of;
        v += (xn - v) * 0.5f;                 // 1/tau
        float sp = (v >= 0.5f) ? 1.f : 0.f;   // spike(v - Vth)
        v -= sp * 0.5f;                       // soft reset
        sa[((size_t)t * L + l) * C + c] = (_Float16)sp;
    }
}

// ============================================================
// Stem convolutions
// ============================================================
__global__ void conv_stem1(const float* __restrict__ x, const float* __restrict__ w,
                           float* __restrict__ y) {
    // x [3][160][64][64], w [32][3][3][3][3], stride (1,2,2), pad 1 -> y [32][160][32][32]
    int i = blockIdx.x * blockDim.x + threadIdx.x;
    if (i >= 32 * 160 * 32 * 32) return;
    int ow = i & 31, oh = (i >> 5) & 31, od = (i >> 10) % 160, oc = i / (160 * 1024);
    float acc = 0.f;
    for (int ic = 0; ic < 3; ++ic)
        for (int kd = 0; kd < 3; ++kd) {
            int id = od + kd - 1; if (id < 0 || id >= 160) continue;
            for (int kh = 0; kh < 3; ++kh) {
                int ih = oh * 2 + kh - 1; if (ih < 0 || ih >= 64) continue;
                for (int kw = 0; kw < 3; ++kw) {
                    int iw = ow * 2 + kw - 1; if (iw < 0 || iw >= 64) continue;
                    acc += x[((size_t)(ic * 160 + id) * 64 + ih) * 64 + iw] *
                           w[(((oc * 3 + ic) * 3 + kd) * 3 + kh) * 3 + kw];
                }
            }
        }
    y[i] = acc;
}

__global__ void conv_stem2(const float* __restrict__ x, const float* __restrict__ w,
                           float* __restrict__ y) {
    // x [32][160][32][32], w [64][32][3][3][3], stride (4,2,2), pad 1 -> y [64][40][16][16]
    int i = blockIdx.x * blockDim.x + threadIdx.x;
    if (i >= 64 * LTOK) return;
    int ow = i & 15, oh = (i >> 4) & 15, od = (i >> 8) % 40, oc = i / LTOK;
    float acc = 0.f;
    for (int ic = 0; ic < 32; ++ic)
        for (int kd = 0; kd < 3; ++kd) {
            int id = od * 4 + kd - 1; if (id < 0 || id >= 160) continue;
            for (int kh = 0; kh < 3; ++kh) {
                int ih = oh * 2 + kh - 1; if (ih < 0 || ih >= 32) continue;
                for (int kw = 0; kw < 3; ++kw) {
                    int iw = ow * 2 + kw - 1; if (iw < 0 || iw >= 32) continue;
                    acc += x[((size_t)(ic * 160 + id) * 32 + ih) * 32 + iw] *
                           w[(((oc * 32 + ic) * 3 + kd) * 3 + kh) * 3 + kw];
                }
            }
        }
    y[i] = acc;
}

__global__ void bn_gelu_inplace(float* __restrict__ y, const float* __restrict__ mean,
                                const float* __restrict__ istd, const float* __restrict__ g,
                                const float* __restrict__ b, int C, int L) {
    int i = blockIdx.x * blockDim.x + threadIdx.x;
    if (i >= C * L) return;
    int c = i / L;
    float v = (y[i] - mean[c]) * istd[c] * g[c] + b[c];
    y[i] = 0.5f * v * (1.f + erff(v * 0.70710678118f));
}

__global__ void stem_finalize(const float* __restrict__ y, const float* __restrict__ mean,
                              const float* __restrict__ istd, const float* __restrict__ g,
                              const float* __restrict__ b, const float* __restrict__ pos,
                              float* __restrict__ xs) {
    int i = blockIdx.x * blockDim.x + threadIdx.x;
    if (i >= C_DIM * LTOK) return;
    int c = i / LTOK, l = i % LTOK, lt = l >> 8;
    float v = (y[i] - mean[c]) * istd[c] * g[c] + b[c] + pos[c * 40 + lt];
    for (int t = 0; t < T_SNN; ++t)
        xs[((size_t)t * C_DIM + c) * LTOK + l] = v;
}

// ============================================================
// Region routing: mean pooled q/k, affinity, top-4
// ============================================================
__global__ void qrkr_mean(const float* __restrict__ qkvw, float* __restrict__ qr,
                          float* __restrict__ kr) {
    int i = blockIdx.x * blockDim.x + threadIdx.x;
    if (i >= T_SNN * WINS * C_DIM) return;
    int c = i & 63, nw = i >> 6;
    const float* base = qkvw + (size_t)nw * SWIN * 192;
    float sq = 0.f, sk = 0.f;
    for (int s = 0; s < SWIN; ++s) { sq += base[s * 192 + c]; sk += base[s * 192 + 64 + c]; }
    qr[i] = sq * (1.f / SWIN);
    kr[i] = sk * (1.f / SWIN);
}

__global__ void __launch_bounds__(64)
route_topk(const float* __restrict__ qr, const float* __restrict__ kr,
           int* __restrict__ idx) {
    int n = blockIdx.x >> 6, i = blockIdx.x & 63, j = threadIdx.x;
    __shared__ float r[64];
    const float* q = qr + (size_t)(n * WINS + i) * C_DIM;
    const float* k = kr + (size_t)(n * WINS + j) * C_DIM;
    float acc = 0.f;
    for (int c = 0; c < C_DIM; ++c) acc += q[c] * k[c];
    r[j] = acc;
    __syncthreads();
    if (j == 0) {
        int chosen[4];
        for (int t = 0; t < 4; ++t) {
            float best = -3.4e38f; int bi = 0;
            for (int u = 0; u < 64; ++u) {
                bool used = false;
                for (int p = 0; p < t; ++p) used = used || (chosen[p] == u);
                if (!used && r[u] > best) { best = r[u]; bi = u; }
            }
            chosen[t] = bi;
            idx[(size_t)(n * WINS + i) * 4 + t] = bi;
        }
    }
}

// ============================================================
// Attention core: per (n,w) workgroup, gathered top-k K/V staged in LDS,
// two-pass fp32 softmax, output fp16 (A layout for out-proj GEMM)
// ============================================================
__global__ void __launch_bounds__(256)
attn_core(const float* __restrict__ qkvw, const int* __restrict__ idx,
          _Float16* __restrict__ ao) {
    extern __shared__ float lds[];
    float* kbuf = lds;                 // [640][16]
    float* vbuf = lds + KGLEN * 16;    // [640][16]
    int nw = blockIdx.x;
    int n = nw >> 6, w = nw & 63;
    int tid = threadIdx.x;
    const int* id4 = idx + (size_t)nw * 4;

    for (int h = 0; h < 4; ++h) {
        __syncthreads();
        for (int e = tid; e < KGLEN * 16; e += 256) {
            int t = e >> 4, d = e & 15;
            int src = id4[t / SWIN];
            int sp  = t % SWIN;
            const float* p = qkvw + ((size_t)(n * WINS + src) * SWIN + sp) * 192;
            kbuf[e] = p[64 + h * 16 + d];
            vbuf[e] = p[128 + h * 16 + d];
        }
        __syncthreads();
        if (tid < SWIN) {
            float qv[16];
            const float* qp = qkvw + ((size_t)nw * SWIN + tid) * 192 + h * 16;
#pragma unroll
            for (int d = 0; d < 16; ++d) qv[d] = qp[d];
            float mx = -3.4e38f;
            for (int t = 0; t < KGLEN; ++t) {
                float s = 0.f;
#pragma unroll
                for (int d = 0; d < 16; ++d) s += qv[d] * kbuf[t * 16 + d];
                mx = fmaxf(mx, s * 0.25f);
            }
            float den = 0.f, acc[16];
#pragma unroll
            for (int d = 0; d < 16; ++d) acc[d] = 0.f;
            for (int t = 0; t < KGLEN; ++t) {
                float s = 0.f;
#pragma unroll
                for (int d = 0; d < 16; ++d) s += qv[d] * kbuf[t * 16 + d];
                s = __expf(s * 0.25f - mx);
                den += s;
#pragma unroll
                for (int d = 0; d < 16; ++d) acc[d] += s * vbuf[t * 16 + d];
            }
            float inv = 1.f / den;
            _Float16* op = ao + ((size_t)n * LTOK + w * SWIN + tid) * C_DIM + h * 16;
#pragma unroll
            for (int d = 0; d < 16; ++d) op[d] = (_Float16)(acc[d] * inv);
        }
    }
}

// ============================================================
// FFN helpers
// ============================================================
__global__ void bn_gelu_tofp16(const float* __restrict__ hbuf, const float* __restrict__ mean,
                               const float* __restrict__ istd, const float* __restrict__ g,
                               const float* __restrict__ b, _Float16* __restrict__ h2) {
    int i = blockIdx.x * blockDim.x + threadIdx.x;
    if (i >= T_SNN * H_DIM * LTOK) return;
    int l = i % LTOK, o = (i / LTOK) % H_DIM, t = i / (H_DIM * LTOK);
    float v = (hbuf[i] - mean[o]) * istd[o] * g[o] + b[o];
    v = 0.5f * v * (1.f + erff(v * 0.70710678118f));
    h2[((size_t)t * LTOK + l) * H_DIM + o] = (_Float16)v;
}

__global__ void bn_residual(float* __restrict__ xs, const float* __restrict__ hout,
                            const float* __restrict__ mean, const float* __restrict__ istd,
                            const float* __restrict__ g, const float* __restrict__ b) {
    int i = blockIdx.x * blockDim.x + threadIdx.x;
    if (i >= T_SNN * C_DIM * LTOK) return;
    int c = (i / LTOK) % C_DIM;
    xs[i] += (hout[i] - mean[c]) * istd[c] * g[c] + b[c];
}

// ============================================================
// Final pooling + interp + layernorm + head
// ============================================================
__global__ void final_pool(const float* __restrict__ xs, float* __restrict__ op) {
    int i = blockIdx.x * blockDim.x + threadIdx.x;
    if (i >= C_DIM * 40) return;
    int c = i / 40, lt = i % 40;
    float s = 0.f;
    for (int t = 0; t < T_SNN; ++t) {
        const float* p = xs + ((size_t)t * C_DIM + c) * LTOK + lt * 256;
        for (int j = 0; j < 256; ++j) s += p[j];
    }
    op[i] = s * (1.f / 1024.f);
}

__global__ void __launch_bounds__(256)
head_kernel(const float* __restrict__ op, const float* __restrict__ ln_g,
            const float* __restrict__ ln_b, const float* __restrict__ hw,
            const float* __restrict__ hb, float* __restrict__ out) {
    __shared__ float rp[256], rs[256];
    int f = threadIdx.x;
    float val = 0.f;
    if (f < 160) {
        float src = (float)f * (39.0f / 159.0f);
        int i0 = (int)floorf(src);
        int i1 = (i0 + 1 < 40) ? (i0 + 1) : 39;
        float wf = src - (float)i0;
        float vec[C_DIM];
        float mu = 0.f;
        for (int c = 0; c < C_DIM; ++c) {
            float v = op[c * 40 + i0] * (1.f - wf) + op[c * 40 + i1] * wf;
            vec[c] = v; mu += v;
        }
        mu *= (1.f / C_DIM);
        float var = 0.f;
        for (int c = 0; c < C_DIM; ++c) { float d = vec[c] - mu; var += d * d; }
        var *= (1.f / C_DIM);
        float rstd = rsqrtf(var + EPSF);
        float acc = hb[0];
        for (int c = 0; c < C_DIM; ++c)
            acc += ((vec[c] - mu) * rstd * ln_g[c] + ln_b[c]) * hw[c];
        val = acc;
    }
    rp[f] = (f < 160) ? val : 0.f;
    rs[f] = rp[f];
    __syncthreads();
    for (int o = 128; o > 0; o >>= 1) {
        if (f < o) rs[f] += rs[f + o];
        __syncthreads();
    }
    float mn = rs[0] / 160.f;
    if (f < 160) out[f] = rp[f] - mn;
}

__global__ void f32_to_f16(const float* __restrict__ src, _Float16* __restrict__ dst, int n) {
    int i = blockIdx.x * blockDim.x + threadIdx.x;
    if (i < n) dst[i] = (_Float16)src[i];
}

// ============================================================
// Host orchestration
// ============================================================
extern "C" void kernel_launch(void* const* d_in, const int* in_sizes, int n_in,
                              void* d_out, int out_size, void* d_ws, size_t ws_size,
                              hipStream_t stream) {
    (void)in_sizes; (void)n_in; (void)out_size; (void)ws_size;

    // Input order: x first, then params tree-flattened with sorted dict keys.
    // Per block (16 entries, sorted; note '1' < '_' < 'o' < 'q' in ASCII):
    //  +0 b1, +1 b2, +2 b3, +3 b4, +4 b_ffn1, +5 b_ffn2, +6 bo, +7 bqkv,
    //  +8 g1, +9 g2, +10 g3, +11 g4, +12 w_ffn1, +13 w_ffn2, +14 wo, +15 wqkv
    // Tail (base 49): head_b, head_w, ln_b, ln_g, pos_embed,
    //                 stem_b1, stem_b2, stem_g1, stem_g2, stem_w1, stem_w2
    const float* x = (const float*)d_in[0];
    auto blk = [&](int b, int k) { return (const float*)d_in[1 + b * 16 + k]; };
    const int TAIL = 1 + 3 * 16;                    // 49
    const float* head_b  = (const float*)d_in[TAIL + 0];
    const float* head_w  = (const float*)d_in[TAIL + 1];
    const float* ln_b    = (const float*)d_in[TAIL + 2];
    const float* ln_g    = (const float*)d_in[TAIL + 3];
    const float* pos     = (const float*)d_in[TAIL + 4];
    const float* stem_b1 = (const float*)d_in[TAIL + 5];
    const float* stem_b2 = (const float*)d_in[TAIL + 6];
    const float* stem_g1 = (const float*)d_in[TAIL + 7];
    const float* stem_g2 = (const float*)d_in[TAIL + 8];
    const float* stem_w1 = (const float*)d_in[TAIL + 9];
    const float* stem_w2 = (const float*)d_in[TAIL + 10];

    // per-block named accessors
    auto p_b1     = [&](int b) { return blk(b, 0); };
    auto p_b2     = [&](int b) { return blk(b, 1); };
    auto p_b3     = [&](int b) { return blk(b, 2); };
    auto p_b4     = [&](int b) { return blk(b, 3); };
    auto p_bffn1  = [&](int b) { return blk(b, 4); };
    auto p_bffn2  = [&](int b) { return blk(b, 5); };
    auto p_bo     = [&](int b) { return blk(b, 6); };
    auto p_bqkv   = [&](int b) { return blk(b, 7); };
    auto p_g1     = [&](int b) { return blk(b, 8); };
    auto p_g2     = [&](int b) { return blk(b, 9); };
    auto p_g3     = [&](int b) { return blk(b, 10); };
    auto p_g4     = [&](int b) { return blk(b, 11); };
    auto p_wffn1  = [&](int b) { return blk(b, 12); };
    auto p_wffn2  = [&](int b) { return blk(b, 13); };
    auto p_wo     = [&](int b) { return blk(b, 14); };
    auto p_wqkv   = [&](int b) { return blk(b, 15); };

    // workspace bump allocator (256B aligned)
    unsigned char* wsb = (unsigned char*)d_ws;
    size_t off = 0;
    auto alloc = [&](size_t bytes) -> void* {
        void* p = wsb + off;
        off += (bytes + 255) & ~(size_t)255;
        return p;
    };
    float*    hbuf  = (float*)alloc((size_t)T_SNN * H_DIM * LTOK * 4);   // also stem1 out
    float*    y2    = (float*)alloc((size_t)C_DIM * LTOK * 4);
    float*    xs    = (float*)alloc((size_t)T_SNN * C_DIM * LTOK * 4);
    _Float16* sa    = (_Float16*)alloc((size_t)M_ROWS * C_DIM * 2);
    float*    qkvw  = (float*)alloc((size_t)T_SNN * WINS * SWIN * 192 * 4);
    float*    qr    = (float*)alloc((size_t)T_SNN * WINS * C_DIM * 4);
    float*    kr    = (float*)alloc((size_t)T_SNN * WINS * C_DIM * 4);
    int*      idxb  = (int*)alloc((size_t)T_SNN * WINS * 4 * 4);
    _Float16* ao    = (_Float16*)alloc((size_t)M_ROWS * C_DIM * 2);
    _Float16* h2a   = (_Float16*)alloc((size_t)M_ROWS * H_DIM * 2);
    float*    hout  = (float*)alloc((size_t)T_SNN * C_DIM * LTOK * 4);
    float*    meanb = (float*)alloc(H_DIM * 4);
    float*    istdb = (float*)alloc(H_DIM * 4);
    float*    opool = (float*)alloc((size_t)C_DIM * 40 * 4);
    _Float16* w16[3][4];
    for (int b = 0; b < 3; ++b) {
        w16[b][0] = (_Float16*)alloc(192 * 64 * 2);   // wqkv
        w16[b][1] = (_Float16*)alloc(64 * 64 * 2);    // wo
        w16[b][2] = (_Float16*)alloc(256 * 64 * 2);   // w_ffn1
        w16[b][3] = (_Float16*)alloc(64 * 256 * 2);   // w_ffn2
    }

    auto cdiv = [](int a, int b) { return (a + b - 1) / b; };

    // ---- stem ----
    conv_stem1<<<cdiv(32 * 160 * 1024, 256), 256, 0, stream>>>(x, stem_w1, hbuf);
    bn_stats<<<32, 256, 0, stream>>>(hbuf, meanb, istdb, 32, 160 * 1024, 1);
    bn_gelu_inplace<<<cdiv(32 * 160 * 1024, 256), 256, 0, stream>>>(hbuf, meanb, istdb, stem_g1, stem_b1, 32, 160 * 1024);
    conv_stem2<<<cdiv(C_DIM * LTOK, 256), 256, 0, stream>>>(hbuf, stem_w2, y2);
    bn_stats<<<C_DIM, 256, 0, stream>>>(y2, meanb, istdb, C_DIM, LTOK, 1);
    stem_finalize<<<cdiv(C_DIM * LTOK, 256), 256, 0, stream>>>(y2, meanb, istdb, stem_g2, stem_b2, pos, xs);

    // ---- weight fp16 conversions ----
    for (int b = 0; b < 3; ++b) {
        f32_to_f16<<<cdiv(192 * 64, 256), 256, 0, stream>>>(p_wqkv(b),  w16[b][0], 192 * 64);
        f32_to_f16<<<cdiv(64 * 64, 256),  256, 0, stream>>>(p_wo(b),    w16[b][1], 64 * 64);
        f32_to_f16<<<cdiv(256 * 64, 256), 256, 0, stream>>>(p_wffn1(b), w16[b][2], 256 * 64);
        f32_to_f16<<<cdiv(64 * 256, 256), 256, 0, stream>>>(p_wffn2(b), w16[b][3], 64 * 256);
    }

    // ---- transformer blocks ----
    for (int b = 0; b < 3; ++b) {
        // BN1 + LIF -> spikes (fp16 GEMM-A layout)
        bn_stats<<<C_DIM, 256, 0, stream>>>(xs, meanb, istdb, C_DIM, LTOK, T_SNN);
        bn_lif<<<cdiv(C_DIM * LTOK, 256), 256, 0, stream>>>(xs, meanb, istdb, p_g1(b), p_b1(b), sa, C_DIM, LTOK);
        // qkv projection (WMMA), store in windowed layout
        {
            int waves = (M_ROWS / 32) * (192 / 32);
            gemm_wmma<1><<<cdiv(waves, 4), 128, 0, stream>>>(sa, w16[b][0], p_bqkv(b), qkvw, nullptr, M_ROWS, 192, 64);
        }
        // routing
        qrkr_mean<<<cdiv(T_SNN * WINS * C_DIM, 256), 256, 0, stream>>>(qkvw, qr, kr);
        route_topk<<<T_SNN * WINS, 64, 0, stream>>>(qr, kr, idxb);
        // attention core (LDS staged gathered K/V, fp32 softmax)
        attn_core<<<T_SNN * WINS, 256, 2 * KGLEN * 16 * sizeof(float), stream>>>(qkvw, idxb, ao);
        // output projection (WMMA) + inverse-window permute + residual into xs
        {
            int waves = (M_ROWS / 32) * (C_DIM / 32);
            gemm_wmma<2><<<cdiv(waves, 4), 128, 0, stream>>>(ao, w16[b][1], p_bo(b), xs, xs, M_ROWS, C_DIM, 64);
        }
        // BN2 + LIF
        bn_stats<<<C_DIM, 256, 0, stream>>>(xs, meanb, istdb, C_DIM, LTOK, T_SNN);
        bn_lif<<<cdiv(C_DIM * LTOK, 256), 256, 0, stream>>>(xs, meanb, istdb, p_g2(b), p_b2(b), sa, C_DIM, LTOK);
        // FFN1 (WMMA) -> hbuf [t][256][L]
        {
            int waves = (M_ROWS / 32) * (H_DIM / 32);
            gemm_wmma<3><<<cdiv(waves, 4), 128, 0, stream>>>(sa, w16[b][2], p_bffn1(b), hbuf, nullptr, M_ROWS, H_DIM, 64);
        }
        // BN3 + GELU -> h2a fp16 [t*L][256]
        bn_stats<<<H_DIM, 256, 0, stream>>>(hbuf, meanb, istdb, H_DIM, LTOK, T_SNN);
        bn_gelu_tofp16<<<cdiv(T_SNN * H_DIM * LTOK, 256), 256, 0, stream>>>(hbuf, meanb, istdb, p_g3(b), p_b3(b), h2a);
        // FFN2 (WMMA) -> hout [t][64][L]
        {
            int waves = (M_ROWS / 32) * (C_DIM / 32);
            gemm_wmma<3><<<cdiv(waves, 4), 128, 0, stream>>>(h2a, w16[b][3], p_bffn2(b), hout, nullptr, M_ROWS, C_DIM, 256);
        }
        // BN4 + residual
        bn_stats<<<C_DIM, 256, 0, stream>>>(hout, meanb, istdb, C_DIM, LTOK, T_SNN);
        bn_residual<<<cdiv(T_SNN * C_DIM * LTOK, 256), 256, 0, stream>>>(xs, hout, meanb, istdb, p_g4(b), p_b4(b));
    }

    // ---- head ----
    final_pool<<<cdiv(C_DIM * 40, 256), 256, 0, stream>>>(xs, opool);
    head_kernel<<<1, 256, 0, stream>>>(opool, ln_g, ln_b, head_w, head_b, (float*)d_out);
}
